// SwinTransformerBlock3D_37649683316965
// MI455X (gfx1250) — matrix operand order repaired and checked by reading, not compiled, not run on hardware
//
#include <hip/hip_runtime.h>
#include <hip/hip_bf16.h>

typedef __attribute__((ext_vector_type(16))) _Float16 v16h;
typedef __attribute__((ext_vector_type(8)))  float    v8f;

// D = A(16x32 f16) * B(32x16 f16) + C(16x16 f32)
__device__ inline v8f wmma16(v16h a, v16h b, v8f c) {
  return __builtin_amdgcn_wmma_f32_16x16x32_f16(false, a, false, b, (short)0, c, false, false);
}

// A-fragment (16x32 f16) from row-major halves with row stride `strideh`.
// Contiguous-k layout used here: lane m=lane&15, k = i + (lane>=16 ? 16 : 0)
__device__ inline v16h load_a_frag(const _Float16* rowbase, int strideh, int k0) {
  int lane = threadIdx.x & 31;
  int m = lane & 15, hi = lane >> 4;
  const _Float16* p = rowbase + m * strideh + k0 + hi * 16;
  v16h a;
#pragma unroll
  for (int i = 0; i < 8; ++i) a[i] = p[i];
#pragma unroll
  for (int i = 0; i < 8; ++i) a[i + 8] = p[8 + i];
  return a;
}

// ---------------- weight packing: W[O][K] row-major f32 -> B fragments ----
// tile(kt,nt): lane l holds 16 contiguous halves, K = kt*32 + i + (l>=16?16:0),
// N = nt*16 + (l&15). B element (k,n) = W[n*K + k]  (B = W^T)
__global__ void pack_b_kernel(const float* __restrict__ Wsrc,
                              _Float16* __restrict__ dst, int Ktot, int Ntot) {
  int ntn = Ntot >> 4;
  int tile = blockIdx.x;
  int kt = tile / ntn, nt = tile % ntn;
  int lane = threadIdx.x;
  int n = nt * 16 + (lane & 15);
  int kbase = kt * 32 + ((lane >> 4) ? 16 : 0);
  _Float16* out = dst + (long)tile * 512 + lane * 16;
#pragma unroll
  for (int i = 0; i < 16; ++i)
    out[i] = (_Float16)Wsrc[(long)n * Ktot + kbase + i];
}

// ---------------- relative position bias gather: [3][98][98] --------------
__global__ void bias3_kernel(const float* __restrict__ tbl,
                             const int* __restrict__ idx, float* __restrict__ out) {
  int t = blockIdx.x * 256 + threadIdx.x;
  if (t >= 3 * 9604) return;
  int h = t / 9604, nm = t % 9604;
  out[t] = tbl[idx[nm] * 3 + h];
}

// ---------------- LN(x2) + roll(-S) + window partition -> f16 Xw ----------
__global__ __launch_bounds__(128) void ln_part_kernel(
    const float* __restrict__ x, const float* __restrict__ gw,
    const float* __restrict__ bw, _Float16* __restrict__ Xw) {
  int win = blockIdx.x;
  int t = threadIdx.x;
  if (t >= 112) return;
  _Float16* outrow = Xw + ((long)win * 112 + t) * 96;
  if (t >= 98) {  // pad rows = exact zeros
    uint4 z = {0u, 0u, 0u, 0u};
#pragma unroll
    for (int i = 0; i < 12; ++i) ((uint4*)outrow)[i] = z;
    return;
  }
  int b = win >> 9, wl = win & 511;
  int wd = wl >> 6, wh = (wl >> 3) & 7, ww = wl & 7;
  int td = t / 49, th = (t / 7) % 7, tw = t % 7;
  int d = wd * 2 + td, h = wh * 7 + th, w = ww * 7 + tw;
  int ds = (d + 1) & 15;
  int hs = h + 3; if (hs >= 56) hs -= 56;
  int wsx = w + 3; if (wsx >= 56) wsx -= 56;
  long base = (((long)(b * 16 + ds) * 56 + hs) * 56 + wsx) * 192 + 96;
  const float4* src4 = (const float4*)(x + base);
  float v[96];
#pragma unroll
  for (int i = 0; i < 24; ++i) {
    float4 f = src4[i];
    v[4 * i] = f.x; v[4 * i + 1] = f.y; v[4 * i + 2] = f.z; v[4 * i + 3] = f.w;
  }
  float s = 0.f, s2 = 0.f;
#pragma unroll
  for (int c = 0; c < 96; ++c) { s += v[c]; s2 += v[c] * v[c]; }
  float mean = s * (1.0f / 96.0f);
  float var = s2 * (1.0f / 96.0f) - mean * mean;
  float inv = rsqrtf(var + 1e-5f);
#pragma unroll
  for (int i = 0; i < 12; ++i) {
    union { _Float16 h8[8]; uint4 u; } pk;
#pragma unroll
    for (int j = 0; j < 8; ++j) {
      int c = i * 8 + j;
      pk.h8[j] = (_Float16)((v[c] - mean) * inv * gw[c] + bw[c]);
    }
    ((uint4*)outrow)[i] = pk.u;
  }
}

// ---------------- QKV GEMM per window: [112,96]x[96,288]+bias -------------
__global__ __launch_bounds__(256) void qkv_kernel(
    const _Float16* __restrict__ Xw, const _Float16* __restrict__ Wp,
    const float* __restrict__ qkvb, _Float16* __restrict__ qg,
    _Float16* __restrict__ kg, _Float16* __restrict__ vt) {
  __shared__ __attribute__((aligned(16))) _Float16 As[112 * 104];
  int win = blockIdx.x;
  int tid = threadIdx.x;
  const uint4* g4 = (const uint4*)(Xw + (long)win * 112 * 96);
  for (int i = tid; i < 112 * 12; i += 256) {
    int row = i / 12, c8 = (i % 12) * 8;
    *(uint4*)&As[row * 104 + c8] = g4[i];
  }
  // zero vT pad columns m in [112,128) -> exact zeros for P@V K padding
  // (two uint4 per (head,dim): 96 * 2 uint4 writes)
  for (int i = tid; i < 192; i += 256) {
    int hd = i >> 1, half = i & 1;
    uint4 z = {0u, 0u, 0u, 0u};
    *(uint4*)(vt + ((long)win * 96 + hd) * 128 + 112 + half * 8) = z;
  }
  __syncthreads();
  int wave = __builtin_amdgcn_readfirstlane(tid >> 5);
  int lane = tid & 31, lcol = lane & 15, hi = lane >> 4;
  for (int t = wave; t < 126; t += 8) {   // 7 Mtiles x 18 Ntiles
    int mt = t / 18, nt = t % 18;
    int n = nt * 16 + lcol;
    float bv = qkvb[n];
    v8f c = {bv, bv, bv, bv, bv, bv, bv, bv};
#pragma unroll
    for (int kt = 0; kt < 3; ++kt) {
      v16h a = load_a_frag(&As[mt * 16 * 104], 104, kt * 32);
      v16h bf = *(const v16h*)(Wp + ((long)(kt * 18 + nt)) * 512 + lane * 16);
      c = wmma16(a, bf, c);
    }
    // tile-uniform q/k/v routing: 288 = [q:0..5][k:6..11][v:12..17] ntiles
    int sec = nt / 6;
    int col = (nt - sec * 6) * 16 + lcol;
    int head = col >> 5, dd = col & 31;
    long base3 = (long)(win * 3 + head);
    int m0 = mt * 16 + hi * 8;
    if (sec == 0) {
      _Float16* dst = qg + (base3 * 112 + m0) * 32 + dd;
#pragma unroll
      for (int r = 0; r < 8; ++r)
        dst[r * 32] = (_Float16)(c[r] * 0.17677669529663687f);
    } else if (sec == 1) {
      _Float16* dst = kg + (base3 * 112 + m0) * 32 + dd;
#pragma unroll
      for (int r = 0; r < 8; ++r) dst[r * 32] = (_Float16)c[r];
    } else {
      // vT[d][m]: r indexes m -> 8 consecutive halves -> one b128 store
      union { _Float16 h8[8]; uint4 u; } pk;
#pragma unroll
      for (int r = 0; r < 8; ++r) pk.h8[r] = (_Float16)c[r];
      *(uint4*)(vt + (base3 * 32 + dd) * 128 + m0) = pk.u;
    }
  }
}

// ---------------- attention per (window, head) ----------------------------
__global__ __launch_bounds__(256) void attn_kernel(
    const _Float16* __restrict__ qg, const _Float16* __restrict__ kg,
    const _Float16* __restrict__ vt, const float* __restrict__ bias3,
    const float* __restrict__ mask, _Float16* __restrict__ attnO) {
  __shared__ __attribute__((aligned(16))) _Float16 P[112 * 136];
  int bid = blockIdx.x;
  int win = bid / 3, head = bid % 3;
  int tid = threadIdx.x;
  int wave = __builtin_amdgcn_readfirstlane(tid >> 5);
  int lane = tid & 31, lcol = lane & 15, hi = lane >> 4;
  // zero P pad columns m in [112,128) (one b128 per row per half)
  for (int i = tid; i < 224; i += 256) {
    int row = i >> 1, half = i & 1;
    uint4 z = {0u, 0u, 0u, 0u};
    *(uint4*)&P[row * 136 + 112 + half * 8] = z;
  }
  long wh = (long)(win * 3 + head);
  const _Float16* qb = qg + wh * 3584;
  const _Float16* kb = kg + wh * 3584;
  const _Float16* vb = vt + wh * 4096;
  const float* mb = mask + (long)(win & 511) * 9604;
  const float* bb = bias3 + head * 9604;

  if (wave < 7) {
    int mt = wave;
    v16h a = load_a_frag(qb + mt * 16 * 32, 32, 0);   // HD=32 -> one K frag
    v8f acc[7];
#pragma unroll
    for (int nt = 0; nt < 7; ++nt) {
      v8f c = {0.f, 0.f, 0.f, 0.f, 0.f, 0.f, 0.f, 0.f};
      // B(k=d, n=m) = K[m][d]: contiguous 16 halves per lane
      v16h bf = *(const v16h*)(kb + (nt * 16 + lcol) * 32 + hi * 16);
      acc[nt] = wmma16(a, bf, c);
    }
#pragma unroll
    for (int nt = 0; nt < 7; ++nt) {
#pragma unroll
      for (int r = 0; r < 8; ++r) {
        int n = mt * 16 + r + hi * 8;
        int m = nt * 16 + lcol;
        float sv = acc[nt][r];
        if (n < 98 && m < 98) sv += bb[n * 98 + m] + mb[n * 98 + m];
        if (m >= 98) sv = -1e9f;
        acc[nt][r] = sv;
      }
    }
#pragma unroll
    for (int r = 0; r < 8; ++r) {
      float mx = acc[0][r];
#pragma unroll
      for (int nt = 1; nt < 7; ++nt) mx = fmaxf(mx, acc[nt][r]);
#pragma unroll
      for (int o = 8; o >= 1; o >>= 1) mx = fmaxf(mx, __shfl_xor(mx, o, 32));
      float sum = 0.f;
#pragma unroll
      for (int nt = 0; nt < 7; ++nt) {
        float p = __expf(acc[nt][r] - mx);
        acc[nt][r] = p;
        sum += p;
      }
#pragma unroll
      for (int o = 8; o >= 1; o >>= 1) sum += __shfl_xor(sum, o, 32);
      float invs = 1.0f / sum;
      int n = mt * 16 + r + hi * 8;
#pragma unroll
      for (int nt = 0; nt < 7; ++nt)
        P[n * 136 + nt * 16 + lcol] = (_Float16)(acc[nt][r] * invs);
    }
  }
  __syncthreads();
  if (wave < 7) {
    int mt = wave;
#pragma unroll
    for (int nt2 = 0; nt2 < 2; ++nt2) {
      v8f c = {0.f, 0.f, 0.f, 0.f, 0.f, 0.f, 0.f, 0.f};
#pragma unroll
      for (int kt = 0; kt < 4; ++kt) {
        v16h a = load_a_frag(&P[mt * 16 * 136], 136, kt * 32);
        // B(k=m, n=d) = vT[d][m]: contiguous 16 halves per lane
        v16h bf = *(const v16h*)(vb + (nt2 * 16 + lcol) * 128 + kt * 32 + hi * 16);
        c = wmma16(a, bf, c);
      }
      int o = head * 32 + nt2 * 16 + lcol;
      _Float16* dst = attnO + ((long)win * 112 + mt * 16 + hi * 8) * 96 + o;
#pragma unroll
      for (int r = 0; r < 8; ++r) dst[r * 96] = (_Float16)c[r];
    }
  }
}

// ---------------- proj GEMM + window_reverse + roll(+S) + x1 residual -----
__global__ __launch_bounds__(256) void proj_kernel(
    const _Float16* __restrict__ attnO, const _Float16* __restrict__ Wp,
    const float* __restrict__ pb, const float* __restrict__ x,
    float* __restrict__ out) {
  __shared__ __attribute__((aligned(16))) _Float16 As[112 * 104];
  int win = blockIdx.x;
  int tid = threadIdx.x;
  const uint4* g4 = (const uint4*)(attnO + (long)win * 112 * 96);
  for (int i = tid; i < 112 * 12; i += 256) {
    int row = i / 12, c8 = (i % 12) * 8;
    *(uint4*)&As[row * 104 + c8] = g4[i];
  }
  __syncthreads();
  int wave = __builtin_amdgcn_readfirstlane(tid >> 5);
  int lane = tid & 31, lcol = lane & 15, hi = lane >> 4;
  int b = win >> 9, wl = win & 511;
  int wd = wl >> 6, whh = (wl >> 3) & 7, www = wl & 7;
  for (int t = wave; t < 42; t += 8) {  // 7 Mtiles x 6 Ntiles
    int mt = t / 6, nt = t % 6;
    int o = nt * 16 + lcol;
    float bv = pb[o];
    v8f c = {bv, bv, bv, bv, bv, bv, bv, bv};
#pragma unroll
    for (int kt = 0; kt < 3; ++kt) {
      v16h a = load_a_frag(&As[mt * 16 * 104], 104, kt * 32);
      v16h bf = *(const v16h*)(Wp + ((long)(kt * 6 + nt)) * 512 + lane * 16);
      c = wmma16(a, bf, c);
    }
#pragma unroll
    for (int r = 0; r < 8; ++r) {
      int n = mt * 16 + r + hi * 8;
      if (n < 98) {
        int td = n / 49, th = (n / 7) % 7, tw = n % 7;
        int d = wd * 2 + td, h = whh * 7 + th, w = www * 7 + tw;
        int dd = (d + 1) & 15;
        int hh2 = h + 3; if (hh2 >= 56) hh2 -= 56;
        int ww2 = w + 3; if (ww2 >= 56) ww2 -= 56;
        long base = (((long)(b * 16 + dd) * 56 + hh2) * 56 + ww2) * 192;
        out[base + o] = x[base + o] + c[r];
      }
    }
  }
}

// ---------------- fused MLP: LN(y1) -> fc1+GELU -> fc2 + x2 residual ------
__global__ __launch_bounds__(256) void mlp_kernel(
    const float* __restrict__ x, const float* __restrict__ g2,
    const float* __restrict__ b2, const _Float16* __restrict__ W1p,
    const float* __restrict__ b1, const _Float16* __restrict__ W2p,
    const float* __restrict__ bf2, float* __restrict__ out) {
  __shared__ __attribute__((aligned(16))) _Float16 G[64 * 104];
  __shared__ __attribute__((aligned(16))) _Float16 Hs[64 * 392];
  long tok0 = (long)blockIdx.x * 64;
  int tid = threadIdx.x;
  if (tid < 64) {  // LayerNorm of y1 (read back from out ch 0..95)
    long tok = tok0 + tid;
    const float4* src4 = (const float4*)(out + tok * 192);
    float v[96];
#pragma unroll
    for (int i = 0; i < 24; ++i) {
      float4 f = src4[i];
      v[4 * i] = f.x; v[4 * i + 1] = f.y; v[4 * i + 2] = f.z; v[4 * i + 3] = f.w;
    }
    float s = 0.f, s2 = 0.f;
#pragma unroll
    for (int c = 0; c < 96; ++c) { s += v[c]; s2 += v[c] * v[c]; }
    float mean = s * (1.0f / 96.0f);
    float var = s2 * (1.0f / 96.0f) - mean * mean;
    float inv = rsqrtf(var + 1e-5f);
#pragma unroll
    for (int i = 0; i < 12; ++i) {
      union { _Float16 h8[8]; uint4 u; } pk;
#pragma unroll
      for (int j = 0; j < 8; ++j) {
        int c = i * 8 + j;
        pk.h8[j] = (_Float16)((v[c] - mean) * inv * g2[c] + b2[c]);
      }
      *(uint4*)&G[tid * 104 + i * 8] = pk.u;
    }
  }
  __syncthreads();
  int wave = __builtin_amdgcn_readfirstlane(tid >> 5);
  int lane = tid & 31, lcol = lane & 15, hi = lane >> 4;
  for (int t = wave; t < 96; t += 8) {  // 4 Mtiles x 24 Ntiles
    int mt = t / 24, nt = t % 24;
    float bv = b1[nt * 16 + lcol];
    v8f c = {bv, bv, bv, bv, bv, bv, bv, bv};
#pragma unroll
    for (int kt = 0; kt < 3; ++kt) {
      v16h a = load_a_frag(&G[mt * 16 * 104], 104, kt * 32);
      v16h bf = *(const v16h*)(W1p + ((long)(kt * 24 + nt)) * 512 + lane * 16);
      c = wmma16(a, bf, c);
    }
#pragma unroll
    for (int r = 0; r < 8; ++r) {
      float vv = c[r];
      float ge = 0.5f * vv * (1.0f + erff(vv * 0.70710678118654752f));
      Hs[(mt * 16 + r + hi * 8) * 392 + nt * 16 + lcol] = (_Float16)ge;
    }
  }
  __syncthreads();
  for (int t = wave; t < 24; t += 8) {  // 4 Mtiles x 6 Ntiles, K = 384
    int mt = t / 6, nt = t % 6;
    int o = nt * 16 + lcol;
    float bv = bf2[o];
    v8f c = {bv, bv, bv, bv, bv, bv, bv, bv};
#pragma unroll
    for (int kt = 0; kt < 12; ++kt) {
      v16h a = load_a_frag(&Hs[mt * 16 * 392], 392, kt * 32);
      v16h bf = *(const v16h*)(W2p + ((long)(kt * 6 + nt)) * 512 + lane * 16);
      c = wmma16(a, bf, c);
    }
#pragma unroll
    for (int r = 0; r < 8; ++r) {
      long tok = tok0 + mt * 16 + r + hi * 8;
      out[tok * 192 + 96 + o] = x[tok * 192 + 96 + o] + c[r];
    }
  }
}

extern "C" void kernel_launch(void* const* d_in, const int* in_sizes, int n_in,
                              void* d_out, int out_size, void* d_ws, size_t ws_size,
                              hipStream_t stream) {
  const float* x      = (const float*)d_in[0];
  const float* mask   = (const float*)d_in[1];
  const int*   relidx = (const int*)d_in[2];
  const float* n1g    = (const float*)d_in[3];
  const float* n1b    = (const float*)d_in[4];
  const float* qkv_w  = (const float*)d_in[5];
  const float* qkv_b  = (const float*)d_in[6];
  const float* rtbl   = (const float*)d_in[7];
  const float* proj_w = (const float*)d_in[8];
  const float* proj_b = (const float*)d_in[9];
  const float* n2g    = (const float*)d_in[10];
  const float* n2b    = (const float*)d_in[11];
  const float* fc1_w  = (const float*)d_in[12];
  const float* fc1_b  = (const float*)d_in[13];
  const float* fc2_w  = (const float*)d_in[14];
  const float* fc2_b  = (const float*)d_in[15];
  float* out = (float*)d_out;
  (void)in_sizes; (void)n_in; (void)out_size; (void)ws_size;

  char* ws = (char*)d_ws;
  size_t off = 0;
  auto take = [&](size_t bytes) {
    char* p = ws + off;
    off = (off + bytes + 255) & ~(size_t)255;
    return p;
  };
  _Float16* qkvWp  = (_Float16*)take(54 * 512 * 2);
  _Float16* projWp = (_Float16*)take(18 * 512 * 2);
  _Float16* fc1Wp  = (_Float16*)take(72 * 512 * 2);
  _Float16* fc2Wp  = (_Float16*)take(72 * 512 * 2);
  float*    bias3  = (float*)take((size_t)3 * 9604 * 4);
  _Float16* Xw     = (_Float16*)take((size_t)1024 * 112 * 96 * 2);
  _Float16* qg     = (_Float16*)take((size_t)1024 * 3 * 112 * 32 * 2);
  _Float16* kg     = (_Float16*)take((size_t)1024 * 3 * 112 * 32 * 2);
  _Float16* vt     = (_Float16*)take((size_t)1024 * 3 * 32 * 128 * 2);
  _Float16* attnO  = (_Float16*)take((size_t)1024 * 112 * 96 * 2);

  pack_b_kernel<<<54, 32, 0, stream>>>(qkv_w, qkvWp, 96, 288);
  pack_b_kernel<<<18, 32, 0, stream>>>(proj_w, projWp, 96, 96);
  pack_b_kernel<<<72, 32, 0, stream>>>(fc1_w, fc1Wp, 96, 384);
  pack_b_kernel<<<72, 32, 0, stream>>>(fc2_w, fc2Wp, 384, 96);
  bias3_kernel<<<(3 * 9604 + 255) / 256, 256, 0, stream>>>(rtbl, relidx, bias3);
  ln_part_kernel<<<1024, 128, 0, stream>>>(x, n1g, n1b, Xw);
  qkv_kernel<<<1024, 256, 0, stream>>>(Xw, qkvWp, qkv_b, qg, kg, vt);
  attn_kernel<<<3072, 256, 0, stream>>>(qg, kg, vt, bias3, mask, attnO);
  proj_kernel<<<1024, 256, 0, stream>>>(attnO, projWp, proj_b, x, out);
  mlp_kernel<<<1568, 256, 0, stream>>>(x, n2g, n2b, fc1Wp, fc1_b, fc2Wp, fc2_b, out);
}
